// MultiHeadAttention_25718264168549
// MI455X (gfx1250) — matrix operand order repaired and checked by reading, not compiled, not run on hardware
//
#include <hip/hip_runtime.h>
#include <hip/hip_bf16.h>

// ---- CDNA5 WMMA types -------------------------------------------------------
typedef _Float16 v8h_t  __attribute__((ext_vector_type(8)));
typedef _Float16 v16h_t __attribute__((ext_vector_type(16)));
typedef float    v8f_t  __attribute__((ext_vector_type(8)));

#define LOG_MAX_SCALE 4.605170185988091f   // log(100)

// A-fragment (16x32 f16, M rows). Lane m (0-15) holds
// K = 0..15; lane m+16 holds K = 16..31. base points at
// (row 0, kBase) of a row-major [rows, ld] f16 array.
static __device__ __forceinline__ v16h_t load_frag_a(const _Float16* base, int ld, int lane) {
    const int m  = lane & 15;
    const int hi = lane >> 4;
    const _Float16* p = base + (size_t)m * ld + hi * 16;
    v8h_t lo = *(const v8h_t*)(p);        // K = kBase + hi*16 .. +7
    v8h_t hh = *(const v8h_t*)(p + 8);    // K = kBase + hi*16 + 8 .. +15
    v16h_t r;
#pragma unroll
    for (int i = 0; i < 8; ++i) { r[i] = lo[i]; r[i + 8] = hh[i]; }
    return r;
}

// B-fragment (32x16 f16, K x N), column-major by N: storage is row-major
// [N, ld] (each "row" is one B-column, contiguous along K). Lane n (0-15)
// holds column n, K=0..15; lane n+16 holds column n, K=16..31.
static __device__ __forceinline__ v16h_t load_frag_b(const _Float16* base, int ld, int lane) {
    const int n  = lane & 15;
    const int hi = lane >> 4;
    const _Float16* p = base + (size_t)n * ld + hi * 16;
    v8h_t a = *(const v8h_t*)(p);
    v8h_t b = *(const v8h_t*)(p + 8);
    v16h_t r;
#pragma unroll
    for (int i = 0; i < 8; ++i) { r[i] = a[i]; r[i + 8] = b[i]; }
    return r;
}

// ---- Prep A: per-token L2 normalization of q and k -> [bh, L, dk] f16 -------
__global__ __launch_bounds__(256) void mha_prep_kernel(
    const float* __restrict__ q, const float* __restrict__ k,
    _Float16* __restrict__ qn, _Float16* __restrict__ kn)
{
    const int col = blockIdx.x * 256 + threadIdx.x;   // [0, 64*1024)
    const int bh  = col >> 10;
    const int t   = col & 1023;
    const size_t base = (size_t)bh * 65536 + t;       // [bh][c][t], c-stride 1024
    float ssq = 0.f, ssk = 0.f;
#pragma unroll 8
    for (int c = 0; c < 64; ++c) { float v = q[base + (size_t)c * 1024]; ssq += v * v; }
#pragma unroll 8
    for (int c = 0; c < 64; ++c) { float v = k[base + (size_t)c * 1024]; ssk += v * v; }
    const float iq = 1.0f / fmaxf(sqrtf(ssq), 1e-12f);
    const float ik = 1.0f / fmaxf(sqrtf(ssk), 1e-12f);
    _Float16* qo = qn + (size_t)col * 64;
    _Float16* ko = kn + (size_t)col * 64;
#pragma unroll 8
    for (int c = 0; c < 64; ++c) {
        qo[c] = (_Float16)(q[base + (size_t)c * 1024] * iq);
        ko[c] = (_Float16)(k[base + (size_t)c * 1024] * ik);
    }
}

// ---- Prep B: raw k -> f16 (acts as V in GEMM2), coalesced 1:1 convert ------
__global__ __launch_bounds__(256) void mha_cvt_kernel(
    const float* __restrict__ k, _Float16* __restrict__ kf)
{
    const size_t gid = (size_t)blockIdx.x * 256 + threadIdx.x;  // [0, 1M)
    const float4 x = ((const float4*)k)[gid];
    v8h_t* out = (v8h_t*)kf;   // pack two float4 -> one 16B store per 2 threads? keep 8B:
    _Float16 h[4] = {(_Float16)x.x, (_Float16)x.y, (_Float16)x.z, (_Float16)x.w};
    // 8-byte store of 4 halves
    typedef _Float16 v4h_t __attribute__((ext_vector_type(4)));
    v4h_t o; o[0]=h[0]; o[1]=h[1]; o[2]=h[2]; o[3]=h[3];
    ((v4h_t*)kf)[gid] = o;
    (void)out;
}

// ---- Fused attention: QK^T (WMMA) -> softmax (LDS) -> attn*V (WMMA) --------
__global__ __launch_bounds__(256) void mha_attn_kernel(
    const _Float16* __restrict__ kf,       // raw k as f16, [bh][c=64][L=1024]
    const float* __restrict__ scale,       // [8] per-head log-scale
    const _Float16* __restrict__ qn,       // [bh][L][64] f16, normalized q^T
    const _Float16* __restrict__ kn,       // [bh][L][64] f16, normalized k^T
    float* __restrict__ out_score,         // [bh][64][1024] f32
    float* __restrict__ out_attn)          // [bh][1024][1024] f32
{
    __shared__ __align__(16) float    lds_logits[16 * 1024];  // 64 KB
    __shared__ __align__(16) _Float16 lds_prob16[16 * 1024];  // 32 KB
    __shared__ __align__(16) float    lds_red[4 * 16 * 16];   //  4 KB

    const int bh    = blockIdx.x >> 6;
    const int tBase = (blockIdx.x & 63) << 4;
    const int tid   = threadIdx.x;
    const int lane  = tid & 31;
    const int wave  = tid >> 5;
    const int head  = bh & 7;
    const float sc  = __expf(fminf(scale[head], LOG_MAX_SCALE));

    // ---------------- GEMM1: logits[t, s] = sc * qn^T kn --------------------
    const _Float16* qrow = qn + ((size_t)bh * 1024 + tBase) * 64;
    const v16h_t a0 = load_frag_a(qrow +  0, 64, lane);   // K = 0..31
    const v16h_t a1 = load_frag_a(qrow + 32, 64, lane);   // K = 32..63

    const int n16 = lane & 15;
    const int hi  = lane >> 4;

#pragma unroll
    for (int i = 0; i < 8; ++i) {
        const int st    = wave * 8 + i;          // s-tile [0,64)
        const int sBase = st << 4;
        const _Float16* krow = kn + ((size_t)bh * 1024 + sBase) * 64;
        v16h_t b0 = load_frag_b(krow +  0, 64, lane);
        v16h_t b1 = load_frag_b(krow + 32, 64, lane);
        v8f_t c = {};
        c = __builtin_amdgcn_wmma_f32_16x16x32_f16(false, a0, false, b0, (short)0, c, false, false);
        c = __builtin_amdgcn_wmma_f32_16x16x32_f16(false, a1, false, b1, (short)0, c, false, false);
        // C layout: VGPR r -> (M = r + hi*8, N = lane&15)
#pragma unroll
        for (int r = 0; r < 8; ++r)
            lds_logits[(r + hi * 8) * 1024 + sBase + n16] = c[r] * sc;
    }
    __syncthreads();

    // ---------------- Softmax over s, write attn (NT), stash f16 probs ------
#pragma unroll
    for (int rr = 0; rr < 2; ++rr) {
        const int row = wave * 2 + rr;
        float* lr = lds_logits + row * 1024;
        float m = -3.0e38f;
        for (int col = lane; col < 1024; col += 32) m = fmaxf(m, lr[col]);
#pragma unroll
        for (int off = 16; off > 0; off >>= 1) m = fmaxf(m, __shfl_xor(m, off, 32));
        float ssum = 0.f;
        for (int col = lane; col < 1024; col += 32) ssum += __expf(lr[col] - m);
#pragma unroll
        for (int off = 16; off > 0; off >>= 1) ssum += __shfl_xor(ssum, off, 32);
        const float inv = 1.0f / ssum;
        float*     oa = out_attn + ((size_t)bh * 1024 + tBase + row) * 1024;
        _Float16*  ph = lds_prob16 + row * 1024;
        for (int col = lane; col < 1024; col += 32) {
            float p = __expf(lr[col] - m) * inv;
            __builtin_nontemporal_store(p, &oa[col]);   // 256MB write-once stream:
            ph[col] = (_Float16)p;                      // keep it out of L2
        }
    }
    __syncthreads();

    // ---------------- GEMM2: score^T[t, c] = attn[t, s] * k^T[s, c] ---------
    // 8 waves = 4 c-tiles x 2 K-halves (K = 1024 over s).
    const int ntile = wave & 3;
    const int khalf = wave >> 2;
    const int cBase = ntile << 4;
    v8f_t acc = {};
#pragma unroll
    for (int ks = 0; ks < 16; ++ks) {
        const int kBase = khalf * 512 + ks * 32;
        v16h_t a = load_frag_a(lds_prob16 + kBase, 1024, lane);
        v16h_t b = load_frag_b(kf + ((size_t)bh * 64 + cBase) * 1024 + kBase, 1024, lane);
        acc = __builtin_amdgcn_wmma_f32_16x16x32_f16(false, a, false, b, (short)0, acc, false, false);
    }
    if (khalf == 1) {
#pragma unroll
        for (int r = 0; r < 8; ++r)
            lds_red[ntile * 256 + (r + hi * 8) * 16 + n16] = acc[r];
    }
    __syncthreads();
    if (khalf == 0) {
#pragma unroll
        for (int r = 0; r < 8; ++r) {
            const int m = r + hi * 8;
            const float v = acc[r] + lds_red[ntile * 256 + m * 16 + n16];
            __builtin_nontemporal_store(v,
                &out_score[((size_t)bh * 64 + cBase + n16) * 1024 + tBase + m]);
        }
    }
}

// ---- Harness entry ----------------------------------------------------------
extern "C" void kernel_launch(void* const* d_in, const int* in_sizes, int n_in,
                              void* d_out, int out_size, void* d_ws, size_t ws_size,
                              hipStream_t stream) {
    const float* q     = (const float*)d_in[0];
    const float* k     = (const float*)d_in[1];
    // d_in[2] (v) is intentionally unused — reference uses k as V.
    const float* scale = (const float*)d_in[3];

    float* out_score = (float*)d_out;                 // 8*512*32*32 = 4,194,304 f32
    float* out_attn  = out_score + (size_t)4194304;   // 8*8*1024*1024 f32

    _Float16* qn = (_Float16*)d_ws;                       // 64*1024*64 f16 = 8 MB
    _Float16* kn = qn + (size_t)64 * 1024 * 64;           // 8 MB
    _Float16* kf = kn + (size_t)64 * 1024 * 64;           // 8 MB (raw k as f16)

    mha_prep_kernel<<<256, 256, 0, stream>>>(q, k, qn, kn);
    mha_cvt_kernel <<<4096, 256, 0, stream>>>(k, kf);     // 4M elems / 4 per thread
    mha_attn_kernel<<<64 * 64, 256, 0, stream>>>(kf, scale, qn, kn, out_score, out_attn);
}
